// MHA_32298154066035
// MI455X (gfx1250) — compile-verified, matrix operand order checked by reading, not run on hardware
//
#include <hip/hip_runtime.h>
#include <hip/hip_bf16.h>

typedef __attribute__((ext_vector_type(16))) _Float16 v16h;
typedef __attribute__((ext_vector_type(8)))  float    v8f;

#define BATCH 2
#define SEQ   2048
#define BDIM  1024
#define NH    16
#define HD    64

// K-index mapping for 16-bit A/B fragments of V_WMMA_*_16X16X32_* (ISA 7.12.2):
// lanes 0-15: elems 0-7 -> K 0-7,  elems 8-15 -> K 16-23
// lanes 16-31: elems 0-7 -> K 8-15, elems 8-15 -> K 24-31
// Per lane this is two contiguous 8-half (16B) runs -> ds_load_b128 x2 when the
// LDS tile is K-major.
__device__ __forceinline__ int kmap(int lane, int e) {
    return ((e >> 3) << 4) + ((lane >> 4) << 3) + (e & 7);
}

// generic->LDS byte offset: flat LDS addresses map by truncation (ISA 10.2)
__device__ __forceinline__ uint32_t lds_off_of(const void* p) {
    return (uint32_t)(uintptr_t)p;
}

// CDNA5 async copy global->LDS, 16B per lane, tracked by ASYNCcnt
__device__ __forceinline__ void async_b128(uint32_t lds_byte_off, const void* g) {
    asm volatile("global_load_async_to_lds_b128 %0, %1, off"
                 :: "v"(lds_byte_off), "v"((unsigned long long)(uintptr_t)g)
                 : "memory");
}
__device__ __forceinline__ void wait_async0() {
    asm volatile("s_wait_asynccnt 0x0" ::: "memory");
}

// ---------------------------------------------------------------- convert f32->f16
__global__ void cvt_f32_f16(const float* __restrict__ in, _Float16* __restrict__ out, int n) {
    int i = blockIdx.x * blockDim.x + threadIdx.x;
    if (i < n) out[i] = (_Float16)in[i];
}

// in[K][N] f32 -> out[N][K] f16  (one-time weight transpose so GEMM B-tiles are K-major)
__global__ void cvt_transpose_f16(const float* __restrict__ in, _Float16* __restrict__ out,
                                  int K, int N) {
    int idx = blockIdx.x * blockDim.x + threadIdx.x;
    if (idx >= K * N) return;
    int k = idx % K, n = idx / K;
    out[(size_t)n * K + k] = (_Float16)in[(size_t)k * N + n];
}

// ---------------------------------------------------------------- WMMA GEMM
// C[M,N] = A[M,K] * Bt[N,K]^T + bias ; f16 in, f32 accum; output f32 (Cf) and/or f16 (Ch)
// 256 threads (8 waves), block tile 128x64, K-step 32, wave tile 32x32,
// double-buffered LDS fed by global_load_async_to_lds_b128.
__global__ __launch_bounds__(256)
void gemm_wmma(const _Float16* __restrict__ A, const _Float16* __restrict__ Bt,
               const float* __restrict__ bias, float* __restrict__ Cf,
               _Float16* __restrict__ Ch, int M, int N, int K)
{
    __shared__ __align__(16) _Float16 As[2][128 * 32];   // [m][k]
    __shared__ __align__(16) _Float16 Bs[2][64 * 32];    // [n][k] (K-major!)

    const int tid  = threadIdx.x;
    const int lane = tid & 31;
    const int wid  = tid >> 5;
    const int waveM = wid & 3;           // 4 waves along M
    const int waveN = wid >> 2;          // 2 waves along N
    const int m0 = blockIdx.y * 128;
    const int n0 = blockIdx.x * 64;
    const int half_lane = lane & 15;
    const int rbase = (lane >> 4) * 8;

    // per-thread staging coordinates (A: 512 uint4 = 2/thread, B: 256 uint4 = 1/thread)
    const int arow0 = tid >> 2,        ac0 = tid & 3;
    const int arow1 = (tid + 256) >> 2, ac1 = (tid + 256) & 3;
    const int brow  = tid >> 2,        bc  = tid & 3;

    auto stage = [&](int buf, int k0) {
        uint32_t abase = lds_off_of(&As[buf][0]);
        uint32_t bbase = lds_off_of(&Bs[buf][0]);
        async_b128(abase + (uint32_t)tid * 16,
                   &A[(size_t)(m0 + arow0) * K + k0 + ac0 * 8]);
        async_b128(abase + (uint32_t)(tid + 256) * 16,
                   &A[(size_t)(m0 + arow1) * K + k0 + ac1 * 8]);
        async_b128(bbase + (uint32_t)tid * 16,
                   &Bt[(size_t)(n0 + brow) * K + k0 + bc * 8]);
    };

    v8f acc[2][2] = {};

    stage(0, 0);
    wait_async0();
    __syncthreads();

    int cur = 0;
    for (int k0 = 0; k0 < K; k0 += 32) {
        if (k0 + 32 < K) stage(cur ^ 1, k0 + 32);   // prefetch next slab (ASYNCcnt)

        v16h af[2], bf[2];
        #pragma unroll
        for (int i = 0; i < 2; ++i) {
            int m = waveM * 32 + i * 16 + half_lane;
            #pragma unroll
            for (int e = 0; e < 16; ++e) af[i][e] = As[cur][m * 32 + kmap(lane, e)];
        }
        #pragma unroll
        for (int j = 0; j < 2; ++j) {
            int n = waveN * 32 + j * 16 + half_lane;
            #pragma unroll
            for (int e = 0; e < 16; ++e) bf[j][e] = Bs[cur][n * 32 + kmap(lane, e)];
        }
        #pragma unroll
        for (int i = 0; i < 2; ++i)
            #pragma unroll
            for (int j = 0; j < 2; ++j)
                acc[i][j] = __builtin_amdgcn_wmma_f32_16x16x32_f16(
                    false, af[i], false, bf[j], (short)0, acc[i][j], false, false);

        wait_async0();       // own async copies into buf^1 landed
        __syncthreads();     // everyone done reading buf (dscnt folded into barrier)
        cur ^= 1;
    }

    #pragma unroll
    for (int i = 0; i < 2; ++i) {
        #pragma unroll
        for (int j = 0; j < 2; ++j) {
            int n = n0 + waveN * 32 + j * 16 + half_lane;
            float bv = bias ? bias[n] : 0.0f;
            #pragma unroll
            for (int r = 0; r < 8; ++r) {
                int m = m0 + waveM * 32 + i * 16 + rbase + r;
                float v = acc[i][j][r] + bv;
                if (Cf) Cf[(size_t)m * N + n] = v;
                if (Ch) Ch[(size_t)m * N + n] = (_Float16)v;
            }
        }
    }
}

// ---------------------------------------------------------------- split heads + present
// qkv f16 [B,S,3D] -> Qh/Kh f16 [B,H,S,HD] (Q pre-scaled 1/sqrt(HD)),
//                     Vt f16 [B,H,HD,S]  (transposed so attention V-tiles are K-major),
//                     present f32 [B,2,H,S,HD]
__global__ void split_heads(const _Float16* __restrict__ qkv,
                            _Float16* __restrict__ Qh, _Float16* __restrict__ Kh,
                            _Float16* __restrict__ Vt, float* __restrict__ present)
{
    int idx = blockIdx.x * blockDim.x + threadIdx.x;   // over B*H*S*HD
    if (idx >= BATCH * NH * SEQ * HD) return;
    int d = idx % HD;
    int s = (idx / HD) % SEQ;
    int h = (idx / (HD * SEQ)) % NH;
    int b = idx / (HD * SEQ * NH);
    size_t src = ((size_t)(b * SEQ + s)) * (3 * BDIM) + h * HD + d;
    float q = (float)qkv[src];
    float k = (float)qkv[src + BDIM];
    float v = (float)qkv[src + 2 * BDIM];
    Qh[idx] = (_Float16)(q * 0.125f);   // rsqrt(64)
    Kh[idx] = (_Float16)k;
    Vt[((size_t)(b * NH + h) * HD + d) * SEQ + s] = (_Float16)v;
    present[((size_t)((b * 2 + 0) * NH + h) * SEQ + s) * HD + d] = k;
    present[((size_t)((b * 2 + 1) * NH + h) * SEQ + s) * HD + d] = v;
}

// ---------------------------------------------------------------- flash attention
// grid (S/64, B*H), 128 threads (4 waves of 16 q-rows), key blocks of 32,
// K/V staged via async-to-LDS; all WMMA fragment reads are ds_load_b128-coalescible.
__global__ __launch_bounds__(128)
void flash_attn(const _Float16* __restrict__ Qh, const _Float16* __restrict__ Kh,
                const _Float16* __restrict__ Vt, _Float16* __restrict__ Ah)
{
    __shared__ __align__(16) _Float16 Ks[32 * 64];      // [key][hd]
    __shared__ __align__(16) _Float16 Vs[64 * 32];      // [hd][key] (K-major for PV)
    __shared__ __align__(16) _Float16 Ps[4][16 * 32];   // per-wave P tile [q][key]

    const int tid  = threadIdx.x;
    const int lane = tid & 31;
    const int wid  = tid >> 5;
    const int half_lane = lane & 15;
    const int rbase = (lane >> 4) * 8;

    const int q0 = blockIdx.x * 64;
    const int bh = blockIdx.y;             // b*NH + h
    const int b = bh / NH, h = bh % NH;
    const size_t base = (size_t)bh * SEQ * HD;
    const int qr = q0 + wid * 16;          // this wave's first q row

    // Q fragments: 16 rows x 64 (two K=32 chunks); contiguous 16B runs per lane
    v16h qf[2];
    #pragma unroll
    for (int f = 0; f < 2; ++f)
        #pragma unroll
        for (int e = 0; e < 16; ++e)
            qf[f][e] = Qh[base + (size_t)(qr + half_lane) * HD + f * 32 + kmap(lane, e)];

    v8f o[4] = {};
    float mi[8], li[8];
    #pragma unroll
    for (int r = 0; r < 8; ++r) { mi[r] = -1e30f; li[r] = 0.0f; }

    const uint32_t ks_base = lds_off_of(&Ks[0]);
    const uint32_t vs_base = lds_off_of(&Vs[0]);

    for (int kb = 0; kb < q0 + 64; kb += 32) {
        // async stage: K block 32x64 (row=key), V block 64x32 (row=hd); 2 uint4 each/thread
        #pragma unroll
        for (int i = 0; i < 2; ++i) {
            int u = tid + i * 128;
            int krow = u >> 3, kc = u & 7;          // Ks: 32 rows x 8 chunks
            async_b128(ks_base + (uint32_t)u * 16,
                       &Kh[base + (size_t)(kb + krow) * HD + kc * 8]);
            int vrow = u >> 2, vc = u & 3;          // Vs: 64 rows x 4 chunks
            async_b128(vs_base + (uint32_t)u * 16,
                       &Vt[base + (size_t)vrow * SEQ + kb + vc * 8]);
        }
        wait_async0();
        __syncthreads();

        // scores: S = Q(16x64) @ K^T(64x32)
        v8f s[2] = {};
        #pragma unroll
        for (int j = 0; j < 2; ++j) {
            #pragma unroll
            for (int f = 0; f < 2; ++f) {
                v16h bf;
                #pragma unroll
                for (int e = 0; e < 16; ++e)
                    bf[e] = Ks[(j * 16 + half_lane) * 64 + f * 32 + kmap(lane, e)];
                s[j] = __builtin_amdgcn_wmma_f32_16x16x32_f16(
                         false, qf[f], false, bf, (short)0, s[j], false, false);
            }
        }

        // causal mask + online softmax (rows live in 16-lane halves -> shfl_xor<=8)
        float mnew[8];
        #pragma unroll
        for (int r = 0; r < 8; ++r) {
            int row = qr + rbase + r;
            #pragma unroll
            for (int j = 0; j < 2; ++j) {
                int key = kb + j * 16 + half_lane;
                if (key > row) s[j][r] = -1e30f;
            }
            float t = fmaxf(s[0][r], s[1][r]);
            #pragma unroll
            for (int off = 8; off >= 1; off >>= 1)
                t = fmaxf(t, __shfl_xor(t, off, 32));
            mnew[r] = fmaxf(mi[r], t);
        }
        #pragma unroll
        for (int r = 0; r < 8; ++r) {
            float scale = __expf(mi[r] - mnew[r]);
            float p0 = __expf(s[0][r] - mnew[r]);
            float p1 = __expf(s[1][r] - mnew[r]);
            s[0][r] = p0; s[1][r] = p1;
            float rs = p0 + p1;
            #pragma unroll
            for (int off = 8; off >= 1; off >>= 1)
                rs += __shfl_xor(rs, off, 32);
            li[r] = li[r] * scale + rs;
            mi[r] = mnew[r];
            #pragma unroll
            for (int t4 = 0; t4 < 4; ++t4) o[t4][r] *= scale;
        }

        // C-layout -> A-layout transpose of P through per-wave LDS tile
        // (intra-wave LDS ordering guaranteed; no cross-wave hazard -> no barrier)
        #pragma unroll
        for (int j = 0; j < 2; ++j)
            #pragma unroll
            for (int r = 0; r < 8; ++r)
                Ps[wid][(rbase + r) * 32 + j * 16 + half_lane] = (_Float16)s[j][r];

        // O += P(16x32) @ V(32x64)
        v16h pf;
        #pragma unroll
        for (int e = 0; e < 16; ++e)
            pf[e] = Ps[wid][half_lane * 32 + kmap(lane, e)];
        #pragma unroll
        for (int t4 = 0; t4 < 4; ++t4) {
            v16h vf;
            #pragma unroll
            for (int e = 0; e < 16; ++e)
                vf[e] = Vs[(t4 * 16 + half_lane) * 32 + kmap(lane, e)];
            o[t4] = __builtin_amdgcn_wmma_f32_16x16x32_f16(
                      false, pf, false, vf, (short)0, o[t4], false, false);
        }
        __syncthreads();   // all waves done reading Ks/Vs before next stage
    }

    // normalize, merge heads -> Ah f16 [B,S,D]
    #pragma unroll
    for (int r = 0; r < 8; ++r) {
        float inv = 1.0f / li[r];
        int row = qr + rbase + r;
        #pragma unroll
        for (int t4 = 0; t4 < 4; ++t4) {
            int col = h * HD + t4 * 16 + half_lane;
            Ah[((size_t)b * SEQ + row) * BDIM + col] = (_Float16)(o[t4][r] * inv);
        }
    }
}

// ----------------------------------------------------------------
extern "C" void kernel_launch(void* const* d_in, const int* in_sizes, int n_in,
                              void* d_out, int out_size, void* d_ws, size_t ws_size,
                              hipStream_t stream)
{
    (void)in_sizes; (void)n_in; (void)out_size; (void)ws_size;
    const float* x      = (const float*)d_in[0];
    const float* w_attn = (const float*)d_in[1];
    const float* b_attn = (const float*)d_in[2];
    const float* w_proj = (const float*)d_in[3];
    const float* b_proj = (const float*)d_in[4];

    float* out_a       = (float*)d_out;                          // [B,S,D]
    float* out_present = out_a + (size_t)BATCH * SEQ * BDIM;     // [B,2,H,S,HD]

    char* ws = (char*)d_ws;
    auto carve = [&](size_t bytes) {
        char* p = ws; ws += (bytes + 255) & ~(size_t)255; return p;
    };
    _Float16* Xh   = (_Float16*)carve(sizeof(_Float16) * (size_t)BATCH * SEQ * BDIM);
    _Float16* Wt   = (_Float16*)carve(sizeof(_Float16) * (size_t)BDIM * 3 * BDIM);  // [3D][D]
    _Float16* Wpt  = (_Float16*)carve(sizeof(_Float16) * (size_t)BDIM * BDIM);      // [D][D]
    _Float16* QKVh = (_Float16*)carve(sizeof(_Float16) * (size_t)BATCH * SEQ * 3 * BDIM);
    _Float16* Qh   = (_Float16*)carve(sizeof(_Float16) * (size_t)BATCH * NH * SEQ * HD);
    _Float16* Kh   = (_Float16*)carve(sizeof(_Float16) * (size_t)BATCH * NH * SEQ * HD);
    _Float16* Vt   = (_Float16*)carve(sizeof(_Float16) * (size_t)BATCH * NH * SEQ * HD);
    _Float16* Ah   = (_Float16*)carve(sizeof(_Float16) * (size_t)BATCH * SEQ * BDIM);

    int n1 = BATCH * SEQ * BDIM;
    cvt_f32_f16<<<(n1 + 255) / 256, 256, 0, stream>>>(x, Xh, n1);
    int n2 = BDIM * 3 * BDIM;
    cvt_transpose_f16<<<(n2 + 255) / 256, 256, 0, stream>>>(w_attn, Wt, BDIM, 3 * BDIM);
    int n3 = BDIM * BDIM;
    cvt_transpose_f16<<<(n3 + 255) / 256, 256, 0, stream>>>(w_proj, Wpt, BDIM, BDIM);

    // QKV projection: [4096x1024] @ [1024x3072] -> f16 QKV
    dim3 g1(3 * BDIM / 64, (BATCH * SEQ) / 128);
    gemm_wmma<<<g1, 256, 0, stream>>>(Xh, Wt, b_attn, nullptr, QKVh,
                                      BATCH * SEQ, 3 * BDIM, BDIM);

    int n4 = BATCH * NH * SEQ * HD;
    split_heads<<<(n4 + 255) / 256, 256, 0, stream>>>(QKVh, Qh, Kh, Vt, out_present);

    dim3 g2(SEQ / 64, BATCH * NH);
    flash_attn<<<g2, 128, 0, stream>>>(Qh, Kh, Vt, Ah);

    // output projection: [4096x1024] @ [1024x1024] + bias -> f32
    dim3 g3(BDIM / 64, (BATCH * SEQ) / 128);
    gemm_wmma<<<g3, 256, 0, stream>>>(Ah, Wpt, b_proj, out_a, nullptr,
                                      BATCH * SEQ, BDIM, BDIM);
}
